// Mamba2Model_55843164782908
// MI455X (gfx1250) — matrix operand order repaired and checked by reading, not compiled, not run on hardware
//
#include <hip/hip_runtime.h>
#include <math.h>

typedef __attribute__((ext_vector_type(16))) _Float16 v16h;
typedef __attribute__((ext_vector_type(8)))  float    v8f;
typedef __attribute__((ext_vector_type(2)))  int      v2i;

#define D_MODEL   256
#define D_INNER   512
#define D_STATE   16
#define NHEADS    8
#define HEADDIM   64
#define CONV_DIM  544        // D_INNER + 2*D_STATE
#define D_IN_PROJ 1064       // 2*D_INNER + 2*D_STATE + NHEADS
#define BATCH     8
#define SEQ       1024       // 32*32 tokens
#define NTOK      8192       // BATCH*SEQ

// --- CDNA5 async global->LDS copy (guarded; falls back to direct LDS store) ---
#if defined(__gfx1250__) && __has_builtin(__builtin_amdgcn_global_load_async_to_lds_b64) && \
    __has_builtin(__builtin_amdgcn_s_wait_asynccnt)
#define ASYNC_COPY_B64(ldsp, gp)                                                   \
  __builtin_amdgcn_global_load_async_to_lds_b64(                                   \
      (__attribute__((address_space(1))) v2i*)(gp),                                \
      (__attribute__((address_space(3))) v2i*)(ldsp), 0, 0)
#define ASYNC_WAIT0() __builtin_amdgcn_s_wait_asynccnt(0)
#else
#define ASYNC_COPY_B64(ldsp, gp)                                                   \
  do { ((float*)(ldsp))[0] = (gp)[0]; ((float*)(ldsp))[1] = (gp)[1]; } while (0)
#define ASYNC_WAIT0() do {} while (0)
#endif

__device__ __forceinline__ float gelu_exact(float v){
  return 0.5f * v * (1.0f + erff(v * 0.70710678118654752f));
}
__device__ __forceinline__ float silu_f(float v){
  return v / (1.0f + expf(-v));
}

// ---------------------------------------------------------------------------
// Weight pre-pack: B (K x N, or N x K if transposed) -> f16 fragment layout
// dst[((k>>5)*N + n)*32 + (k&31)], so a WMMA B fragment is one 32B v16h load.
// ---------------------------------------------------------------------------
__global__ void pack_b_kernel(const float* __restrict__ src, _Float16* __restrict__ dst,
                              int K, int N, int transposed)
{
  size_t idx = (size_t)blockIdx.x * 256 + threadIdx.x;
  if (idx >= (size_t)K * N) return;
  int k = (int)(idx / N);
  int n = (int)(idx % N);
  float v = transposed ? src[(size_t)n * K + k] : src[(size_t)k * N + n];
  dst[((size_t)(k >> 5) * N + n) * 32 + (k & 31)] = (_Float16)v;
}

// ---------------------------------------------------------------------------
// Stem conv1: 3x3 SAME, 3->128 ch, + scale/bias + exact GELU.  NCHW.
// ---------------------------------------------------------------------------
__global__ void stem1_kernel(const float* __restrict__ x, const float* __restrict__ w,
                             const float* __restrict__ g, const float* __restrict__ bia,
                             float* __restrict__ h1)
{
  size_t idx = (size_t)blockIdx.x * 256 + threadIdx.x;   // 8*128*128*128
  int xx = idx & 127;
  int yy = (idx >> 7) & 127;
  int co = (idx >> 14) & 127;
  int bb = (int)(idx >> 21);
  float s = 0.0f;
  for (int ci = 0; ci < 3; ++ci) {
    for (int ky = 0; ky < 3; ++ky) {
      int iy = yy + ky - 1;
      if (iy < 0 || iy > 127) continue;
      for (int kx = 0; kx < 3; ++kx) {
        int ix = xx + kx - 1;
        if (ix < 0 || ix > 127) continue;
        s += x[(((size_t)bb * 3 + ci) * 128 + iy) * 128 + ix] *
             w[((co * 3 + ci) * 3 + ky) * 3 + kx];
      }
    }
  }
  s = s * g[co] + bia[co];
  h1[idx] = gelu_exact(s);
}

// ---------------------------------------------------------------------------
// Stem conv2 as implicit-im2col WMMA GEMM: M=8192 (b*oy*ox), N=256, K=2048
// (k = ci*16 + ky*4 + kx).  Weights pre-packed f16.  Epilogue: bn + GELU.
// ---------------------------------------------------------------------------
__global__ void conv2_wmma_kernel(const float* __restrict__ h1,
                                  const _Float16* __restrict__ w2p,
                                  const float* __restrict__ g2, const float* __restrict__ b2,
                                  float* __restrict__ tok)
{
  int wave = threadIdx.x >> 5;
  int lane = threadIdx.x & 31;
  int n0 = (blockIdx.x * 8 + wave) * 16;
  int m0 = blockIdx.y * 16;
  int lh = lane >> 4;
  int mr = m0 + (lane & 15);
  int bb = mr >> 10;
  int l  = mr & 1023;
  int oy = l >> 5, ox = l & 31;
  size_t abase = (size_t)bb * 128 * 16384 + (size_t)(oy * 4) * 128 + (size_t)(ox * 4);
  int n = n0 + (lane & 15);
  v8f acc = {};
  for (int kt = 0; kt < 64; ++kt) {
    int kb = kt * 32;
    v16h a;
#pragma unroll
    for (int i = 0; i < 16; ++i) {
      int k = kb + lh * 16 + i;
      int ci = k >> 4, ky = (k >> 2) & 3, kx = k & 3;
      a[i] = (_Float16)h1[abase + (size_t)ci * 16384 + ky * 128 + kx];
    }
    v16h bv = *(const v16h*)(w2p + ((size_t)kt * 256 + n) * 32 + lh * 16);
    acc = __builtin_amdgcn_wmma_f32_16x16x32_f16(false, a, false, bv, (short)0, acc,
                                                 false, false);
  }
  float gg = g2[n], bbias = b2[n];
#pragma unroll
  for (int v = 0; v < 8; ++v) {
    int mm = m0 + lh * 8 + v;
    float val = acc[v] * gg + bbias;
    tok[(size_t)mm * D_MODEL + n] = gelu_exact(val);
  }
}

// ---------------------------------------------------------------------------
// WMMA GEMM: C[M,N] = A[M,K] * Bp (pre-packed f16 fragments).
// 8 waves/block share one 16-row M tile staged in LDS with double-buffered
// async global->LDS copies; each wave owns one 16-col N tile.
// ---------------------------------------------------------------------------
__global__ void gemm_wmma_kernel(const float* __restrict__ A, const _Float16* __restrict__ Bp,
                                 float* __restrict__ C, int N, int K,
                                 int lda, int ldc)
{
  __shared__ float At[2][16][32];
  int wave = threadIdx.x >> 5;
  int lane = threadIdx.x & 31;
  int n0 = (blockIdx.x * 8 + wave) * 16;
  int m0 = blockIdx.y * 16;
  int lh = lane >> 4;
  int ml = lane & 15;
  int n  = n0 + ml;
  int ncl = (n < N) ? n : 0;          // clamp loads; stores masked

  // cooperative A-tile copy: 256 threads x 8B = 2KB tile
  int t  = threadIdx.x;
  int cr = t >> 4;                    // row 0..15
  int cc = (t & 15) * 2;              // col 0..30
  const float* Abase = A + (size_t)(m0 + cr) * lda + cc;

  ASYNC_COPY_B64(&At[0][cr][cc], Abase);          // prologue: k-block 0
  v8f acc = {};
  int nkt = K >> 5;
  for (int kt = 0; kt < nkt; ++kt) {
    int buf = kt & 1;
    ASYNC_WAIT0();                    // my slice of current buffer landed
    __syncthreads();                  // everyone's slice landed / prev compute done
    if (kt + 1 < nkt)
      ASYNC_COPY_B64(&At[buf ^ 1][cr][cc], Abase + (kt + 1) * 32);
    v16h a;
    const float* Ar = &At[buf][ml][lh * 16];
#pragma unroll
    for (int i = 0; i < 8; ++i) {
      a[i]     = (_Float16)Ar[i];
      a[i + 8] = (_Float16)Ar[8 + i];
    }
    v16h bv = *(const v16h*)(Bp + ((size_t)kt * N + ncl) * 32 + lh * 16);
    acc = __builtin_amdgcn_wmma_f32_16x16x32_f16(false, a, false, bv, (short)0, acc,
                                                 false, false);
  }
  if (n < N) {
#pragma unroll
    for (int v = 0; v < 8; ++v) {
      int mm = m0 + lh * 8 + v;
      C[(size_t)mm * ldc + n] = acc[v];
    }
  }
}

// ---------------------------------------------------------------------------
// Depthwise causal conv (width 4) over xBC channels + bias + SiLU.
// zx row layout: [z(512) | xBC(544) | dt(8)], stride 1064.
// ---------------------------------------------------------------------------
__global__ void dwconv_silu_kernel(const float* __restrict__ zx, const float* __restrict__ cw,
                                   const float* __restrict__ cb, float* __restrict__ xbc)
{
  int idx = blockIdx.x * 256 + threadIdx.x;   // 8192*544
  int c = idx % CONV_DIM;
  int m = idx / CONV_DIM;
  int bb = m >> 10, l = m & 1023;
  float s = cb[c];
#pragma unroll
  for (int k = 0; k < 4; ++k) {
    int ll = l + k - 3;
    if (ll >= 0)
      s += zx[(size_t)((bb << 10) + ll) * D_IN_PROJ + D_INNER + c] * cw[c * 4 + k];
  }
  xbc[(size_t)m * CONV_DIM + c] = silu_f(s);
}

// ---------------------------------------------------------------------------
// dt = softplus(raw + dt_bias); dA = exp(-exp(A_log) * dt).   8192*8 elems.
// ---------------------------------------------------------------------------
__global__ void dtda_kernel(const float* __restrict__ zx, const float* __restrict__ dt_bias,
                            const float* __restrict__ A_log, float* __restrict__ dtb,
                            float* __restrict__ dab)
{
  int idx = blockIdx.x * 256 + threadIdx.x;   // 65536
  int h = idx & 7;
  int m = idx >> 3;
  float v = zx[(size_t)m * D_IN_PROJ + (D_IN_PROJ - NHEADS) + h] + dt_bias[h];
  float sp = (v > 20.0f) ? v : log1pf(expf(v));
  dtb[idx] = sp;
  dab[idx] = expf(-expf(A_log[h]) * sp);
}

// ---------------------------------------------------------------------------
// Selective scan: one block per (batch, head), 64 threads = head dim p.
// State row (16 wide) in registers; no barriers, 32 FMA/step/thread.
// ---------------------------------------------------------------------------
__global__ void scan_kernel(const float* __restrict__ xbc, const float* __restrict__ dtb,
                            const float* __restrict__ dab, const float* __restrict__ Dsk,
                            float* __restrict__ yb)
{
  int bb = blockIdx.x >> 3;
  int h  = blockIdx.x & 7;
  int p  = threadIdx.x;               // 0..63
  float st[D_STATE];
#pragma unroll
  for (int n = 0; n < D_STATE; ++n) st[n] = 0.0f;
  float Dh = Dsk[h];
  for (int l = 0; l < SEQ; ++l) {
    int m = (bb << 10) + l;
    float a   = dab[m * NHEADS + h];
    float dtv = dtb[m * NHEADS + h];
    const float* row = xbc + (size_t)m * CONV_DIM;
    float xv  = row[h * HEADDIM + p];
    float dtx = xv * dtv;
    float y   = xv * Dh;
#pragma unroll
    for (int n = 0; n < D_STATE; ++n) {
      st[n] = st[n] * a + dtx * row[D_INNER + n];
      y += st[n] * row[D_INNER + D_STATE + n];
    }
    yb[(size_t)m * D_INNER + h * HEADDIM + p] = y;
  }
}

// ---------------------------------------------------------------------------
// y *= silu(z); RMS-norm over 512 with rms_w.  One block (256 thr) per token.
// ---------------------------------------------------------------------------
__global__ void gate_rms_kernel(float* __restrict__ yb, const float* __restrict__ zx,
                                const float* __restrict__ rw)
{
  int m = blockIdx.x;
  int t = threadIdx.x;
  __shared__ float red[256];
  float* yrow = yb + (size_t)m * D_INNER;
  const float* zrow = zx + (size_t)m * D_IN_PROJ;
  float vals[2];
  float ss = 0.0f;
#pragma unroll
  for (int j = 0; j < 2; ++j) {
    int d = t + j * 256;
    float z = zrow[d];
    float y = yrow[d] * silu_f(z);
    vals[j] = y;
    ss += y * y;
  }
  red[t] = ss; __syncthreads();
  for (int s = 128; s > 0; s >>= 1) { if (t < s) red[t] += red[t + s]; __syncthreads(); }
  float r = rsqrtf(red[0] * (1.0f / 512.0f) + 1e-5f);
#pragma unroll
  for (int j = 0; j < 2; ++j) {
    int d = t + j * 256;
    yrow[d] = vals[j] * r * rw[d];
  }
}

// ---------------------------------------------------------------------------
// tok = LayerNorm(m + tok) over 256.  One block per token.
// ---------------------------------------------------------------------------
__global__ void add_ln_kernel(float* __restrict__ tok, const float* __restrict__ mb,
                              const float* __restrict__ g, const float* __restrict__ bta)
{
  int m = blockIdx.x;
  int t = threadIdx.x;
  __shared__ float red[256];
  size_t base = (size_t)m * D_MODEL;
  float v = tok[base + t] + mb[base + t];
  red[t] = v; __syncthreads();
  for (int s = 128; s > 0; s >>= 1) { if (t < s) red[t] += red[t + s]; __syncthreads(); }
  float mu = red[0] * (1.0f / 256.0f); __syncthreads();
  float d = v - mu;
  red[t] = d * d; __syncthreads();
  for (int s = 128; s > 0; s >>= 1) { if (t < s) red[t] += red[t + s]; __syncthreads(); }
  float var = red[0] * (1.0f / 256.0f);
  tok[base + t] = d * rsqrtf(var + 1e-5f) * g[t] + bta[t];
}

// ---------------------------------------------------------------------------
// Mean pool over L.  2048 threads (b,c).
// ---------------------------------------------------------------------------
__global__ void pool_kernel(const float* __restrict__ tok, float* __restrict__ pooled)
{
  int idx = blockIdx.x * 256 + threadIdx.x;   // 2048
  int bb = idx >> 8, c = idx & 255;
  float s = 0.0f;
  for (int l = 0; l < SEQ; ++l)
    s += tok[(((size_t)bb << 10) + l) * D_MODEL + c];
  pooled[idx] = s * (1.0f / 1024.0f);
}

// ---------------------------------------------------------------------------
// Head: LN(pooled) @ head_w + head_b.  One block per batch.
// ---------------------------------------------------------------------------
__global__ void head_kernel(const float* __restrict__ pooled, const float* __restrict__ g,
                            const float* __restrict__ bta, const float* __restrict__ hw,
                            const float* __restrict__ hb, float* __restrict__ out)
{
  int bb = blockIdx.x;
  int t  = threadIdx.x;
  __shared__ float red[256];
  __shared__ float nv[256];
  float v = pooled[bb * 256 + t];
  red[t] = v; __syncthreads();
  for (int s = 128; s > 0; s >>= 1) { if (t < s) red[t] += red[t + s]; __syncthreads(); }
  float mu = red[0] * (1.0f / 256.0f); __syncthreads();
  float d = v - mu;
  red[t] = d * d; __syncthreads();
  for (int s = 128; s > 0; s >>= 1) { if (t < s) red[t] += red[t + s]; __syncthreads(); }
  float var = red[0] * (1.0f / 256.0f);
  nv[t] = d * rsqrtf(var + 1e-5f) * g[t] + bta[t];
  __syncthreads();
  if (t < 10) {
    float s = hb[t];
    for (int k = 0; k < 256; ++k) s += nv[k] * hw[k * 10 + t];
    out[bb * 10 + t] = s;
  }
}

// ---------------------------------------------------------------------------
// Workspace layout (floats).  Region A (stem h1, 16,777,216 floats) is reused
// per layer for zxbcdt / xBC / dt / dA after conv2 consumes h1.
// ---------------------------------------------------------------------------
#define H1_OFF    ((size_t)0)
#define ZX_OFF    ((size_t)0)                      // 8192*1064 = 8,716,288
#define XBC_OFF   ((size_t)8716288)                // 8192*544  = 4,456,448
#define DT_OFF    ((size_t)(8716288 + 4456448))    // 65,536
#define DA_OFF    (DT_OFF + 65536)                 // ends 13,303,808 < 16,777,216
#define TOK_OFF   ((size_t)16777216)               // 2,097,152
#define Y_OFF     (TOK_OFF + 2097152)              // 4,194,304
#define MB_OFF    (Y_OFF + 4194304)                // 2,097,152
#define POOL_OFF  (MB_OFF + 2097152)               // 2,048
#define PACK_OFF  (POOL_OFF + 2048)                // f16 packed weights region
// half-element offsets inside PACK region:
#define W2P_HOFF   ((size_t)0)                     // 2048*256  = 524,288 halfs
#define INWP_HOFF  ((size_t)524288)                // 4 * 272,384
#define OUTWP_HOFF ((size_t)(524288 + 4*272384))   // 4 * 131,072

extern "C" void kernel_launch(void* const* d_in, const int* in_sizes, int n_in,
                              void* d_out, int out_size, void* d_ws, size_t ws_size,
                              hipStream_t stream)
{
  const float* x        = (const float*)d_in[0];
  const float* stem_w1  = (const float*)d_in[1];
  const float* stem_g1  = (const float*)d_in[2];
  const float* stem_b1  = (const float*)d_in[3];
  const float* stem_w2  = (const float*)d_in[4];
  const float* stem_g2  = (const float*)d_in[5];
  const float* stem_b2  = (const float*)d_in[6];
  const float* in_w     = (const float*)d_in[7];
  const float* conv_w   = (const float*)d_in[8];
  const float* conv_b   = (const float*)d_in[9];
  const float* dt_bias  = (const float*)d_in[10];
  const float* A_log    = (const float*)d_in[11];
  const float* D_skip   = (const float*)d_in[12];
  const float* rms_w    = (const float*)d_in[13];
  const float* out_w    = (const float*)d_in[14];
  const float* ln_g     = (const float*)d_in[15];
  const float* ln_b     = (const float*)d_in[16];
  const float* head_ln_g= (const float*)d_in[17];
  const float* head_ln_b= (const float*)d_in[18];
  const float* head_w   = (const float*)d_in[19];
  const float* head_b   = (const float*)d_in[20];
  (void)in_sizes; (void)n_in; (void)out_size; (void)ws_size;

  float* ws   = (float*)d_ws;
  float* h1   = ws + H1_OFF;
  float* zx   = ws + ZX_OFF;
  float* xbc  = ws + XBC_OFF;
  float* dtb  = ws + DT_OFF;
  float* dab  = ws + DA_OFF;
  float* tok  = ws + TOK_OFF;
  float* yb   = ws + Y_OFF;
  float* mb   = ws + MB_OFF;
  float* pool = ws + POOL_OFF;
  _Float16* packbase = (_Float16*)(ws + PACK_OFF);
  _Float16* w2p   = packbase + W2P_HOFF;
  _Float16* inwp  = packbase + INWP_HOFF;
  _Float16* outwp = packbase + OUTWP_HOFF;
  float* out  = (float*)d_out;

  // Weight pre-pack (f16 fragment layout)
  pack_b_kernel<<<2048, 256, 0, stream>>>(stem_w2, w2p, 2048, 256, 1);
  for (int i = 0; i < 4; ++i) {
    pack_b_kernel<<<1064, 256, 0, stream>>>(in_w + (size_t)i * D_MODEL * D_IN_PROJ,
                                            inwp + (size_t)i * 272384, D_MODEL, D_IN_PROJ, 0);
    pack_b_kernel<<<512, 256, 0, stream>>>(out_w + (size_t)i * D_INNER * D_MODEL,
                                           outwp + (size_t)i * 131072, D_INNER, D_MODEL, 0);
  }

  // Stem
  stem1_kernel<<<65536, 256, 0, stream>>>(x, stem_w1, stem_g1, stem_b1, h1);
  conv2_wmma_kernel<<<dim3(2, 512), 256, 0, stream>>>(h1, w2p, stem_g2, stem_b2, tok);

  // 4 Mamba2 layers
  for (int i = 0; i < 4; ++i) {
    // zxbcdt = tok @ in_w[i]  (M=8192, N=1064, K=256): 67 N-tiles -> grid.x=9
    gemm_wmma_kernel<<<dim3(9, 512), 256, 0, stream>>>(
        tok, inwp + (size_t)i * 272384, zx, D_IN_PROJ, D_MODEL, D_MODEL, D_IN_PROJ);
    dwconv_silu_kernel<<<17408, 256, 0, stream>>>(
        zx, conv_w + (size_t)i * CONV_DIM * 4, conv_b + (size_t)i * CONV_DIM, xbc);
    dtda_kernel<<<256, 256, 0, stream>>>(zx, dt_bias + i * NHEADS, A_log + i * NHEADS,
                                         dtb, dab);
    scan_kernel<<<64, 64, 0, stream>>>(xbc, dtb, dab, D_skip + i * NHEADS, yb);
    gate_rms_kernel<<<NTOK, 256, 0, stream>>>(yb, zx, rms_w + (size_t)i * D_INNER);
    // m = y @ out_w[i]  (M=8192, N=256, K=512): 16 N-tiles -> grid.x=2
    gemm_wmma_kernel<<<dim3(2, 512), 256, 0, stream>>>(
        yb, outwp + (size_t)i * 131072, mb, D_MODEL, D_INNER, D_INNER, D_MODEL);
    add_ln_kernel<<<NTOK, 256, 0, stream>>>(tok, mb, ln_g + i * D_MODEL,
                                            ln_b + i * D_MODEL);
  }

  // Pool + head
  pool_kernel<<<8, 256, 0, stream>>>(tok, pool);
  head_kernel<<<8, 256, 0, stream>>>(pool, head_ln_g, head_ln_b, head_w, head_b, out);
}
